// LinearAttentionPathway_4011499454829
// MI455X (gfx1250) — matrix operand order repaired and checked by reading, not compiled, not run on hardware
//
#include <hip/hip_runtime.h>

// ---------------------------------------------------------------------------
// Problem constants (match the reference)
// ---------------------------------------------------------------------------
#define Bn 2
#define Tn 2048
#define Dn 1024
#define Fn 2048            // RFF feature dim = 2*Dn
#define BTn (Bn * Tn)
#define RFF_SCALE 0.044194173824159216f   // sqrt(2/1024)
#define EPSv 1e-6f
#define LDSP 40            // padded LDS row stride (u16) for 32-wide k tiles

typedef unsigned short u16;
typedef __attribute__((ext_vector_type(16))) __bf16 v16bf;
typedef __attribute__((ext_vector_type(8)))  float  v8f;

union Frag { v16bf v; uint4 q[2]; };

__device__ __forceinline__ u16 f2bf(float x) {
  unsigned u = __float_as_uint(x);
  return (u16)((u + 0x7FFFu + ((u >> 16) & 1u)) >> 16);   // RNE
}
__device__ __forceinline__ float bf2f(u16 h) {
  return __uint_as_float(((unsigned)h) << 16);
}

// --- WMMA fragment loads from LDS (CDNA5 wave32 layouts, CDNA5 ISA) -------
// A 16x32 bf16: lane L -> row m = L%16; k runs [(L/16)*8 .. +7] and [+16 ..]
__device__ __forceinline__ void load_frag_a(const u16* s, int lane, int m0, Frag& f) {
  const u16* p = s + (m0 + (lane & 15)) * LDSP + ((lane >> 4) << 3);
  f.q[0] = *(const uint4*)(p);
  f.q[1] = *(const uint4*)(p + 16);
}
// B 32x16 bf16: lane L -> col n = L%16; k runs [(L/16)*8 .. +7] and [+16 ..]
__device__ __forceinline__ void load_frag_b(const u16* s, int lane, int n0, Frag& f) {
  const u16* p = s + (n0 + (lane & 15)) * LDSP + ((lane >> 4) << 3);
  f.q[0] = *(const uint4*)(p);
  f.q[1] = *(const uint4*)(p + 16);
}
__device__ __forceinline__ v8f wmma_bf16(const Frag& a, const Frag& b, v8f c) {
  return __builtin_amdgcn_wmma_f32_16x16x32_bf16(false, a.v, false, b.v,
                                                 (short)0, c, false, false);
}

// --- staging: exact-trip unrolled copies, 16B vector moves ------------------
// 64 rows x 32 k of u16, 128 threads: 256 16B-groups -> 2 iterations exact.
__device__ __forceinline__ void stage64x32_u16(const u16* __restrict__ g, long ldg,
                                               u16* __restrict__ s, int tid) {
#pragma unroll
  for (int it = 0; it < 2; ++it) {
    int idx = tid + it * 128;
    int r = idx >> 2, c = (idx & 3) << 3;
    *(uint4*)(s + r * LDSP + c) = *(const uint4*)(g + (long)r * ldg + c);
  }
}
// 64 rows x 32 k of f32 -> bf16 in LDS, 128 threads: 512 4-elem groups -> 4 it.
__device__ __forceinline__ void stage64x32_f32bf(const float* __restrict__ g, long ldg,
                                                 u16* __restrict__ s, int tid) {
#pragma unroll
  for (int it = 0; it < 4; ++it) {
    int idx = tid + it * 128;
    int r = idx >> 3, c = (idx & 7) << 2;
    float4 f = *(const float4*)(g + (long)r * ldg + c);
    uint2 o;
    o.x = (unsigned)f2bf(f.x) | ((unsigned)f2bf(f.y) << 16);
    o.y = (unsigned)f2bf(f.z) | ((unsigned)f2bf(f.w) << 16);
    *(uint2*)(s + r * LDSP + c) = o;
  }
}
// 128 rows x 32 k of u16, 256 threads: 512 16B-groups -> 2 iterations exact.
__device__ __forceinline__ void stage128x32_u16(const u16* __restrict__ g, long ldg,
                                                u16* __restrict__ s, int tid) {
#pragma unroll
  for (int it = 0; it < 2; ++it) {
    int idx = tid + it * 256;
    int r = idx >> 2, c = (idx & 3) << 3;
    *(uint4*)(s + r * LDSP + c) = *(const uint4*)(g + (long)r * ldg + c);
  }
}

#define WAVE_SETUP_4()                                  \
  int tid = threadIdx.x, lane = tid & 31;               \
  int wave = tid >> 5;                                  \
  int wm = (wave & 1) << 5, wn = (wave >> 1) << 5

#define DO_WMMA_STEP(sAp, sBp)                          \
  do {                                                  \
    Frag a0, a1, b0, b1;                                \
    load_frag_a(sAp, lane, wm,      a0);                \
    load_frag_a(sAp, lane, wm + 16, a1);                \
    load_frag_b(sBp, lane, wn,      b0);                \
    load_frag_b(sBp, lane, wn + 16, b1);                \
    acc[0][0] = wmma_bf16(a0, b0, acc[0][0]);           \
    acc[0][1] = wmma_bf16(a0, b1, acc[0][1]);           \
    acc[1][0] = wmma_bf16(a1, b0, acc[1][0]);           \
    acc[1][1] = wmma_bf16(a1, b1, acc[1][1]);           \
  } while (0)

// ---------------------------------------------------------------------------
// Main bf16 GEMM: C(MxN) = A(MxK, row-major) * B(KxN), Bt = B^T row-major.
// 128x128 WG tile, 256 threads (8 wave32, 4x2), double-buffered LDS:
// 8 WMMAs per wave per K=32 step, one barrier per step, loads overlap compute.
// ---------------------------------------------------------------------------
__global__ __launch_bounds__(256) void gemm_bf16_kernel(
    const u16* __restrict__ A, const u16* __restrict__ Bt,
    float* __restrict__ Cf, u16* __restrict__ Cbf, int M, int N, int K) {
  __shared__ u16 sA[2][128 * LDSP];
  __shared__ u16 sB[2][128 * LDSP];
  int tid = threadIdx.x, lane = tid & 31, wave = tid >> 5;
  int wm = (wave & 3) << 5;          // 0,32,64,96
  int wn = (wave >> 2) << 6;         // 0,64
  long m0 = (long)blockIdx.x * 128, n0 = (long)blockIdx.y * 128;
  v8f acc[2][4] = {};

  stage128x32_u16(A  + m0 * K, K, sA[0], tid);
  stage128x32_u16(Bt + n0 * K, K, sB[0], tid);
  __syncthreads();
  const int nk = K >> 5;
  for (int ks = 0; ks < nk; ++ks) {
    int cur = ks & 1, nxt = cur ^ 1;
    if (ks + 1 < nk) {
      stage128x32_u16(A  + m0 * K + (long)(ks + 1) * 32, K, sA[nxt], tid);
      stage128x32_u16(Bt + n0 * K + (long)(ks + 1) * 32, K, sB[nxt], tid);
    }
    Frag a0, a1, b0, b1, b2, b3;
    load_frag_a(sA[cur], lane, wm,      a0);
    load_frag_a(sA[cur], lane, wm + 16, a1);
    load_frag_b(sB[cur], lane, wn,      b0);
    load_frag_b(sB[cur], lane, wn + 16, b1);
    load_frag_b(sB[cur], lane, wn + 32, b2);
    load_frag_b(sB[cur], lane, wn + 48, b3);
    acc[0][0] = wmma_bf16(a0, b0, acc[0][0]);
    acc[0][1] = wmma_bf16(a0, b1, acc[0][1]);
    acc[0][2] = wmma_bf16(a0, b2, acc[0][2]);
    acc[0][3] = wmma_bf16(a0, b3, acc[0][3]);
    acc[1][0] = wmma_bf16(a1, b0, acc[1][0]);
    acc[1][1] = wmma_bf16(a1, b1, acc[1][1]);
    acc[1][2] = wmma_bf16(a1, b2, acc[1][2]);
    acc[1][3] = wmma_bf16(a1, b3, acc[1][3]);
    __syncthreads();
  }
#pragma unroll
  for (int i = 0; i < 2; ++i)
#pragma unroll
    for (int j = 0; j < 4; ++j)
#pragma unroll
      for (int r = 0; r < 8; ++r) {
        long gm = m0 + wm + i * 16 + ((lane >> 4) << 3) + r;
        long gn = n0 + wn + j * 16 + (lane & 15);
        float v = acc[i][j][r];
        if (Cf)  Cf[gm * N + gn]  = v;
        if (Cbf) Cbf[gm * N + gn] = f2bf(v);
      }
}

// ---------------------------------------------------------------------------
// Chunked linear attention (chunk C = 64); 64x64 tiles, 128 threads, 4 waves
// ---------------------------------------------------------------------------
// A_c = causal_mask(Qphi_c @ Kphi_c^T)  -> bf16 (64x64 per batch)
__global__ __launch_bounds__(128) void attn_scores_kernel(
    const u16* __restrict__ Qphi, const u16* __restrict__ Kphi,
    u16* __restrict__ Ac, int chunk) {
  __shared__ u16 sA[2][64 * LDSP];
  __shared__ u16 sB[2][64 * LDSP];
  WAVE_SETUP_4();
  int b = blockIdx.x;
  const u16* Ab = Qphi + ((long)b * Tn + (long)chunk * 64) * Fn;
  const u16* Bb = Kphi + ((long)b * Tn + (long)chunk * 64) * Fn;  // Kphi^T col-major == Kphi row-major
  v8f acc[2][2] = {};
  stage64x32_u16(Ab, Fn, sA[0], tid);
  stage64x32_u16(Bb, Fn, sB[0], tid);
  __syncthreads();
  const int nk = Fn >> 5;
  for (int ks = 0; ks < nk; ++ks) {
    int cur = ks & 1, nxt = cur ^ 1;
    if (ks + 1 < nk) {
      stage64x32_u16(Ab + (long)(ks + 1) * 32, Fn, sA[nxt], tid);
      stage64x32_u16(Bb + (long)(ks + 1) * 32, Fn, sB[nxt], tid);
    }
    DO_WMMA_STEP(sA[cur], sB[cur]);
    __syncthreads();
  }
#pragma unroll
  for (int i = 0; i < 2; ++i)
#pragma unroll
    for (int j = 0; j < 2; ++j)
#pragma unroll
      for (int r = 0; r < 8; ++r) {
        int tq = wm + i * 16 + ((lane >> 4) << 3) + r;
        int tk = wn + j * 16 + (lane & 15);
        float v = (tk <= tq) ? acc[i][j][r] : 0.f;   // inclusive causal mask
        Ac[((long)b << 12) + (long)tq * 64 + tk] = f2bf(v);
      }
}

// gamma_c = (Qphi_c @ S_prev + A_c @ V_c) / (den + eps); S stored as S^T (Dv x F)
__global__ __launch_bounds__(128) void attn_out_kernel(
    const u16* __restrict__ Qphi, const u16* __restrict__ Ac,
    const u16* __restrict__ Vt, const float* __restrict__ St,
    const float* __restrict__ den, u16* __restrict__ ybf, int chunk) {
  __shared__ u16 sA[2][64 * LDSP];
  __shared__ u16 sB[2][64 * LDSP];
  WAVE_SETUP_4();
  int b = blockIdx.y;
  long dv0 = (long)blockIdx.x * 64;
  v8f acc[2][2] = {};
  // inter-chunk: Qphi_c (64xF) @ S (FxDv); B col-major == rows of S^T (f32->bf16)
  const u16*   Ab = Qphi + ((long)b * Tn + (long)chunk * 64) * Fn;
  const float* Sb = St + ((long)b * Dn + dv0) * Fn;
  stage64x32_u16(Ab, Fn, sA[0], tid);
  stage64x32_f32bf(Sb, Fn, sB[0], tid);
  __syncthreads();
  const int nk = Fn >> 5;
  for (int ks = 0; ks < nk; ++ks) {
    int cur = ks & 1, nxt = cur ^ 1;
    if (ks + 1 < nk) {
      stage64x32_u16(Ab + (long)(ks + 1) * 32, Fn, sA[nxt], tid);
      stage64x32_f32bf(Sb + (long)(ks + 1) * 32, Fn, sB[nxt], tid);
    }
    DO_WMMA_STEP(sA[cur], sB[cur]);
    __syncthreads();
  }
  // intra-chunk: A_c (64x64) @ V_c (64xDv); B col-major == rows of V^T
  const u16* Ab2 = Ac + ((long)b << 12);
  const u16* Vb  = Vt + ((long)b * Dn + dv0) * Tn + (long)chunk * 64;
#pragma unroll
  for (int k0 = 0; k0 < 64; k0 += 32) {
    stage64x32_u16(Ab2 + k0, 64, sA[0], tid);
    stage64x32_u16(Vb  + k0, Tn, sB[0], tid);
    __syncthreads();
    DO_WMMA_STEP(sA[0], sB[0]);
    __syncthreads();
  }
#pragma unroll
  for (int i = 0; i < 2; ++i)
#pragma unroll
    for (int j = 0; j < 2; ++j)
#pragma unroll
      for (int r = 0; r < 8; ++r) {
        long t  = (long)chunk * 64 + wm + i * 16 + ((lane >> 4) << 3) + r;
        long gn = dv0 + wn + j * 16 + (lane & 15);
        float d = den[(long)b * Tn + t];
        float g = acc[i][j][r] / (d + EPSv);
        ybf[((long)b * Tn + t) * Dn + gn] = f2bf(g);
      }
}

// S^T (Dv x F) += V_c^T (Dv x 64) @ Kphi_c (64 x F); B col-major == rows of Kphi^T
__global__ __launch_bounds__(128) void state_update_kernel(
    const u16* __restrict__ Vt, const u16* __restrict__ Kphit,
    float* __restrict__ St, int chunk) {
  __shared__ u16 sA[2][64 * LDSP];
  __shared__ u16 sB[2][64 * LDSP];
  WAVE_SETUP_4();
  int b = blockIdx.z;
  long dv0 = (long)blockIdx.x * 64, f0 = (long)blockIdx.y * 64;
  const u16* Ab = Vt    + ((long)b * Dn + dv0) * Tn + (long)chunk * 64;
  const u16* Bb = Kphit + ((long)b * Fn + f0)  * Tn + (long)chunk * 64;
  v8f acc[2][2] = {};
#pragma unroll
  for (int k0 = 0; k0 < 64; k0 += 32) {
    stage64x32_u16(Ab + k0, Tn, sA[0], tid);
    stage64x32_u16(Bb + k0, Tn, sB[0], tid);
    __syncthreads();
    DO_WMMA_STEP(sA[0], sB[0]);
    __syncthreads();
  }
#pragma unroll
  for (int i = 0; i < 2; ++i)
#pragma unroll
    for (int j = 0; j < 2; ++j)
#pragma unroll
      for (int r = 0; r < 8; ++r) {
        long gm = dv0 + wm + i * 16 + ((lane >> 4) << 3) + r;  // dv
        long gn = f0  + wn + j * 16 + (lane & 15);             // f
        St[((long)b * Dn + gm) * Fn + gn] += acc[i][j][r];
      }
}

// ---------------------------------------------------------------------------
// Elementwise / helper kernels
// ---------------------------------------------------------------------------
__global__ void cvt_f32_bf16_kernel(const float* __restrict__ s, u16* __restrict__ d, long n) {
  long i = (long)blockIdx.x * blockDim.x + threadIdx.x;
  if (i < n) d[i] = f2bf(s[i]);
}
// W (RxC f32) -> W^T (CxR bf16)
__global__ void cvt_transpose_f32_bf16_kernel(const float* __restrict__ s, u16* __restrict__ d,
                                              int rows, int cols) {
  long i = (long)blockIdx.x * blockDim.x + threadIdx.x;
  if (i < (long)rows * cols) {
    long r = i / cols, c = i % cols;
    d[c * rows + r] = f2bf(s[i]);
  }
}
// batched u16 transpose: src[b][r][c] -> dst[b][c][r]
__global__ void transpose_u16_kernel(const u16* __restrict__ s, u16* __restrict__ d,
                                     int rows, int cols) {
  int b = blockIdx.z;
  long i = (long)blockIdx.x * blockDim.x + threadIdx.x;
  if (i < (long)rows * cols) {
    long r = i / cols, c = i % cols;
    d[((long)b * cols + c) * rows + r] = s[((long)b * rows + r) * cols + c];
  }
}
__global__ void zero_f32_kernel(float* __restrict__ p, long n) {
  long i = (long)blockIdx.x * blockDim.x + threadIdx.x;
  if (i < n) p[i] = 0.f;
}
// phi = scale*[cos(P), sin(P)]; optionally also write phi^T (per batch F x T)
__global__ void rff_kernel(const float* __restrict__ proj, u16* __restrict__ phi,
                           u16* __restrict__ phit) {
  long i = (long)blockIdx.x * blockDim.x + threadIdx.x;
  if (i >= (long)Bn * Tn * Dn) return;
  long b = i / ((long)Tn * Dn), r = i % ((long)Tn * Dn);
  long t = r / Dn, dd = r % Dn;
  float p = proj[i];
  u16 cb = f2bf(RFF_SCALE * __cosf(p));
  u16 sb = f2bf(RFF_SCALE * __sinf(p));
  long base = (b * Tn + t) * Fn;
  phi[base + dd]      = cb;
  phi[base + Dn + dd] = sb;
  if (phit) {
    phit[(b * Fn + dd)      * Tn + t] = cb;
    phit[(b * Fn + Dn + dd) * Tn + t] = sb;
  }
}
// inclusive cumsum of Kphi over t, per (b, f) lane
__global__ void cumsum_kernel(const u16* __restrict__ kphi, float* __restrict__ cum) {
  long g = (long)blockIdx.x * blockDim.x + threadIdx.x;
  if (g >= (long)Bn * Fn) return;
  long b = g / Fn, f = g % Fn;
  float run = 0.f;
  for (long t = 0; t < Tn; ++t) {
    long idx = (b * Tn + t) * Fn + f;
    run += bf2f(kphi[idx]);
    cum[idx] = run;
  }
}
// den[b,t] = sum_f qphi[b,t,f] * cumK[b,t,f]  (one wave32 per row)
__global__ void den_kernel(const u16* __restrict__ qphi, const float* __restrict__ cum,
                           float* __restrict__ den) {
  int lane = threadIdx.x & 31;
  long row = (long)blockIdx.x * 8 + (threadIdx.x >> 5);
  if (row >= (long)Bn * Tn) return;
  const u16*   qp = qphi + row * Fn;
  const float* cp = cum + row * Fn;
  float a = 0.f;
  for (int f = lane; f < Fn; f += 32) a += bf2f(qp[f]) * cp[f];
  for (int off = 16; off > 0; off >>= 1) a += __shfl_down(a, off, 32);
  if (lane == 0) den[row] = a;
}

// ---------------------------------------------------------------------------
// Host orchestration
// ---------------------------------------------------------------------------
extern "C" void kernel_launch(void* const* d_in, const int* in_sizes, int n_in,
                              void* d_out, int out_size, void* d_ws, size_t ws_size,
                              hipStream_t stream) {
  (void)in_sizes; (void)n_in; (void)out_size; (void)ws_size;
  const float* x  = (const float*)d_in[0];
  const float* Wq = (const float*)d_in[1];
  const float* Wk = (const float*)d_in[2];
  const float* Wv = (const float*)d_in[3];
  const float* Wo = (const float*)d_in[4];
  const float* Om = (const float*)d_in[5];

  char* w = (char*)d_ws;
  size_t off = 0;
  auto alloc = [&](size_t bytes) -> char* {
    off = (off + 255) & ~(size_t)255;
    char* p = w + off; off += bytes; return p;
  };
  const size_t BTD = (size_t)Bn * Tn * Dn;
  const size_t BTF = (size_t)Bn * Tn * Fn;
  const size_t DD  = (size_t)Dn * Dn;

  u16*  xbf   = (u16*)alloc(BTD * 2);
  u16*  wqt   = (u16*)alloc(DD * 2);
  u16*  wkt   = (u16*)alloc(DD * 2);
  u16*  wvt   = (u16*)alloc(DD * 2);
  u16*  wot   = (u16*)alloc(DD * 2);
  u16*  omt   = (u16*)alloc(DD * 2);
  u16*  qbf   = (u16*)alloc(BTD * 2);
  u16*  kbf   = (u16*)alloc(BTD * 2);
  u16*  vbf   = (u16*)alloc(BTD * 2);
  u16*  vt    = (u16*)alloc(BTD * 2);
  float* proj = (float*)alloc(BTD * 4);
  u16*  qphi  = (u16*)alloc(BTF * 2);
  u16*  kphi  = (u16*)alloc(BTF * 2);
  u16*  kphit = (u16*)alloc(BTF * 2);
  float* cum  = (float*)alloc(BTF * 4);
  float* den  = (float*)alloc((size_t)Bn * Tn * 4);
  u16*  ac    = (u16*)alloc((size_t)Bn * 64 * 64 * 2);
  float* st   = (float*)alloc((size_t)Bn * Dn * Fn * 4);
  u16*  ybf   = (u16*)alloc(BTD * 2);

  const int THR = 256;
  auto nblk = [](long n, int t) { return (unsigned)((n + t - 1) / t); };

  // 1) bf16 copies (weights transposed so every GEMM B-operand is col-major)
  cvt_f32_bf16_kernel<<<nblk(BTD, THR), THR, 0, stream>>>(x, xbf, (long)BTD);
  cvt_transpose_f32_bf16_kernel<<<nblk(DD, THR), THR, 0, stream>>>(Wq, wqt, Dn, Dn);
  cvt_transpose_f32_bf16_kernel<<<nblk(DD, THR), THR, 0, stream>>>(Wk, wkt, Dn, Dn);
  cvt_transpose_f32_bf16_kernel<<<nblk(DD, THR), THR, 0, stream>>>(Wv, wvt, Dn, Dn);
  cvt_transpose_f32_bf16_kernel<<<nblk(DD, THR), THR, 0, stream>>>(Wo, wot, Dn, Dn);
  cvt_transpose_f32_bf16_kernel<<<nblk(DD, THR), THR, 0, stream>>>(Om, omt, Dn, Dn);

  // 2) Q/K/V projections (WMMA GEMMs, 128x128 tiles)
  dim3 gP(BTn / 128, Dn / 128);
  gemm_bf16_kernel<<<gP, 256, 0, stream>>>(xbf, wqt, nullptr, qbf, BTn, Dn, Dn);
  gemm_bf16_kernel<<<gP, 256, 0, stream>>>(xbf, wkt, nullptr, kbf, BTn, Dn, Dn);
  gemm_bf16_kernel<<<gP, 256, 0, stream>>>(xbf, wvt, nullptr, vbf, BTn, Dn, Dn);
  transpose_u16_kernel<<<dim3(nblk((long)Tn * Dn, THR), 1, Bn), THR, 0, stream>>>(vbf, vt, Tn, Dn);

  // 3) RFF features (proj buffer reused for Q then K)
  gemm_bf16_kernel<<<gP, 256, 0, stream>>>(qbf, omt, proj, nullptr, BTn, Dn, Dn);
  rff_kernel<<<nblk(BTD, THR), THR, 0, stream>>>(proj, qphi, nullptr);
  gemm_bf16_kernel<<<gP, 256, 0, stream>>>(kbf, omt, proj, nullptr, BTn, Dn, Dn);
  rff_kernel<<<nblk(BTD, THR), THR, 0, stream>>>(proj, kphi, kphit);

  // 4) denominator hoisted out of the chunk loop
  cumsum_kernel<<<nblk((long)Bn * Fn, THR), THR, 0, stream>>>(kphi, cum);
  den_kernel<<<nblk((long)Bn * Tn, 8), THR, 0, stream>>>(qphi, cum, den);

  // 5) chunked linear attention; S^T lives in ws (fits in 192MB L2)
  zero_f32_kernel<<<nblk((long)Bn * Dn * Fn, THR), THR, 0, stream>>>(st, (long)Bn * Dn * Fn);
  for (int c = 0; c < Tn / 64; ++c) {
    attn_scores_kernel<<<dim3(Bn), 128, 0, stream>>>(qphi, kphi, ac, c);
    attn_out_kernel<<<dim3(Dn / 64, Bn), 128, 0, stream>>>(qphi, ac, vt, st, den, ybf, c);
    state_update_kernel<<<dim3(Dn / 64, Fn / 64, Bn), 128, 0, stream>>>(vt, kphit, st, c);
  }

  // 6) output projection -> f32 d_out
  gemm_bf16_kernel<<<gP, 256, 0, stream>>>(ybf, wot, (float*)d_out, nullptr, BTn, Dn, Dn);
}
